// VAEAKFCombinedLinear_4123168604713
// MI455X (gfx1250) — compile-verified, matrix operand order checked by reading, not run on hardware
//
#include <hip/hip_runtime.h>

// ---------------------------------------------------------------------------
// Types for CDNA5 WMMA (wave32)
// ---------------------------------------------------------------------------
typedef __bf16 bf16_t;
typedef __attribute__((ext_vector_type(16))) bf16_t v16bf;
typedef __attribute__((ext_vector_type(8)))  float  v8f;

union FragU { v16bf v; unsigned int u[8]; };

__device__ __forceinline__ unsigned short f2bf(float x) {
  unsigned u = __float_as_uint(x);
  u += 0x7fffu + ((u >> 16) & 1u);            // round-to-nearest-even
  return (unsigned short)(u >> 16);
}
__device__ __forceinline__ float bf2f(unsigned short h) {
  return __uint_as_float((unsigned)h << 16);
}
__device__ __forceinline__ float sigmoidf(float x) { return 1.0f / (1.0f + expf(-x)); }

__device__ __forceinline__ unsigned hash_u32(unsigned x) {
  x ^= x >> 16; x *= 0x7feb352du; x ^= x >> 15; x *= 0x846ca68bu; x ^= x >> 16;
  return x;
}
// Deterministic standard normal (stand-in for JAX threefry normal)
__device__ __forceinline__ float hash_normal(unsigned idx, unsigned salt) {
  unsigned h1 = hash_u32(idx * 2654435761u + salt);
  unsigned h2 = hash_u32(h1 ^ 0x9e3779b9u);
  float u1 = (float)(h1 >> 8) * (1.0f / 16777216.0f) + 1e-7f;
  float u2 = (float)(h2 >> 8) * (1.0f / 16777216.0f);
  return sqrtf(-2.0f * logf(u1)) * cosf(6.28318530718f * u2);
}

// ---------------------------------------------------------------------------
// CDNA5 async global->LDS copy (GLOBAL_LOAD_ASYNC_TO_LDS_B128, ASYNCcnt).
// lds generic pointer: low 32 bits are the wave-relative LDS byte address.
// ---------------------------------------------------------------------------
__device__ __forceinline__ void async_ld_lds_b128(void* lds, const void* g) {
  unsigned off = (unsigned)(size_t)lds;
  asm volatile("global_load_async_to_lds_b128 %0, %1, off"
               :: "v"(off), "v"(g) : "memory");
}
__device__ __forceinline__ void wait_async0() {
  asm volatile("s_wait_asynccnt 0x0" ::: "memory");
}
__device__ __forceinline__ void wait_async2() {   // allow newest batch (2) in flight
  asm volatile("s_wait_asynccnt 0x2" ::: "memory");
}
__device__ __forceinline__ void wait_async4() {   // allow newest batch (4) in flight
  asm volatile("s_wait_asynccnt 0x4" ::: "memory");
}

// 16-bit A-style fragment loader, CDNA5 ISA 7.12.2 layout.
// Row-major source, tile = 16 rows x 32 K. Also serves as the B fragment for
// D = X * W^T when W is (out,in) row-major (lane = output column = W row).
__device__ __forceinline__ v16bf ld_frag(const unsigned short* p, int ld) {
  int lane = threadIdx.x & 31;
  const unsigned short* row = p + (lane & 15) * ld + (lane >> 4) * 8;
  FragU f;
#pragma unroll
  for (int v = 0; v < 4; ++v) {
    f.u[v]     = *(const unsigned int*)(row + 2 * v);        // K 0..7  (+hi*8)
    f.u[v + 4] = *(const unsigned int*)(row + 16 + 2 * v);   // K 16..23(+hi*8)
  }
  return f.v;
}

// Same layout, converting from an fp32 row-major source on the fly.
__device__ __forceinline__ v16bf ld_frag_f32(const float* p, int ld) {
  int lane = threadIdx.x & 31;
  const float* row = p + (lane & 15) * ld + (lane >> 4) * 8;
  FragU f;
#pragma unroll
  for (int v = 0; v < 8; ++v) {
    int k = (v < 4) ? 2 * v : 16 + 2 * (v - 4);
    f.u[v] = (unsigned)f2bf(row[k]) | ((unsigned)f2bf(row[k + 1]) << 16);
  }
  return f.v;
}

__device__ __forceinline__ v8f wmma_bf16(v16bf a, v16bf b, v8f c) {
  return __builtin_amdgcn_wmma_f32_16x16x32_bf16(false, a, false, b, (short)0, c,
                                                 false, false);
}

#define VZERO8 {0.f,0.f,0.f,0.f,0.f,0.f,0.f,0.f}

// ---------------------------------------------------------------------------
// Kernel: fp32 -> bf16 weight conversion
// ---------------------------------------------------------------------------
__global__ void k_cvt_bf16(const float* __restrict__ s, unsigned short* __restrict__ d, int n) {
  int i = blockIdx.x * blockDim.x + threadIdx.x;
  if (i < n) d[i] = f2bf(s[i]);
}

// ---------------------------------------------------------------------------
// Kernel: u = tanh(ext @ u_W^T + u_b)   (L*B,32)x(32->256), bf16 out
// ---------------------------------------------------------------------------
__global__ void k_embed_u(const float* __restrict__ ext,
                          const unsigned short* __restrict__ uW,   // (256,32) bf16
                          const float* __restrict__ ub,
                          unsigned short* __restrict__ u_bf) {
  const v8f vz = VZERO8;
  int row0 = blockIdx.x * 16;
  int wave = threadIdx.x >> 5, lane = threadIdx.x & 31;
  int n = lane & 15, hi = lane >> 4;
  v16bf a = ld_frag_f32(ext + (size_t)row0 * 32, 32);   // K = 32 exactly
  for (int ct = wave; ct < 16; ct += 4) {
    v8f acc = wmma_bf16(a, ld_frag(uW + ct * 16 * 32, 32), vz);
    int col = ct * 16 + n;
#pragma unroll
    for (int v = 0; v < 8; ++v) {
      int r = row0 + v + 8 * hi;
      u_bf[(size_t)r * 256 + col] = f2bf(tanhf(acc[v] + ub[col]));
    }
  }
}

// ---------------------------------------------------------------------------
// Kernel: state0 = mu0 + exp(ls0)*eps from gauss_decoder DBlock on h0
// ---------------------------------------------------------------------------
__global__ void k_state0(const float* __restrict__ h0,
                         const float* __restrict__ W1, const float* __restrict__ b1,
                         const float* __restrict__ W2, const float* __restrict__ b2,
                         const float* __restrict__ Wm, const float* __restrict__ bm,
                         const float* __restrict__ Ws, const float* __restrict__ bs,
                         float* __restrict__ state0) {
  __shared__ float s_t[512];
  __shared__ float s_mu[64];
  __shared__ float s_ls[64];
  int tid = threadIdx.x;  // 256
  for (int j = tid; j < 512; j += 256) {
    float a1 = b1[j], a2 = b2[j];
    for (int k = 0; k < 256; ++k) {
      float h = h0[k];
      a1 += h * W1[j * 256 + k];
      a2 += h * W2[j * 256 + k];
    }
    s_t[j] = tanhf(a1) * sigmoidf(a2);
  }
  __syncthreads();
  if (tid < 128) {
    int o = tid & 63;
    const float* W = (tid < 64) ? Wm : Ws;
    float acc = (tid < 64) ? bm[o] : bs[o];
    for (int k = 0; k < 512; ++k) acc += s_t[k] * W[o * 512 + k];
    if (tid < 64) s_mu[o] = acc; else s_ls[o] = acc;
  }
  __syncthreads();
  for (int idx = tid; idx < 512 * 64; idx += 256) {
    int o = idx & 63;
    state0[idx] = s_mu[o] + expf(s_ls[o]) * hash_normal((unsigned)idx, 0x1111u);
  }
}

// ---------------------------------------------------------------------------
// Kernel: dynamics scan. One persistent block per 16 batch rows; 256 threads
// (8 wave32s); loops t = 0..255 with recurrent state held in LDS.
// u_t is double-buffered and fetched with async global->LDS one step ahead.
// ---------------------------------------------------------------------------
__global__ void k_dynamics(const float* __restrict__ state0,
                           const float* __restrict__ w0,
                           const unsigned short* __restrict__ u_bf,
                           const unsigned short* __restrict__ Wih,   // (1024,64)  bf16
                           const unsigned short* __restrict__ Whh,   // (1024,256) bf16
                           const float* __restrict__ bias,           // 1024
                           const float* __restrict__ lwW,            // (8,256) fp32
                           const float* __restrict__ lwb,            // 8
                           const unsigned short* __restrict__ linW,  // (512,320) bf16
                           const float* __restrict__ linb,           // 512 (flat 8x64)
                           const float* __restrict__ logsig,         // 64
                           float* __restrict__ stf,
                           unsigned short* __restrict__ stb) {
  extern __shared__ char smem[];
  float*          s_gates = (float*)smem;                         // 16x1024 = 64KB
  float*          s_cand  = (float*)(smem + 65536);               // 16x512  = 32KB
  float*          s_wc    = (float*)(smem + 98304);               // 16x256  = 16KB
  unsigned short* s_u     = (unsigned short*)(smem + 114688);     // 2x16x256= 16KB
  unsigned short* s_state = (unsigned short*)(smem + 131072);     // 16x64   = 2KB
  unsigned short* s_whbf  = (unsigned short*)(smem + 133120);     // 16x256  = 8KB
  float*          s_w     = (float*)(smem + 141312);              // 16x8
  float*          s_lw    = (float*)(smem + 141824);              // 8x256   = 8KB
  float*          s_bias  = (float*)(smem + 150016);              // 1024    = 4KB
  float*          s_linb  = (float*)(smem + 154112);              // 512     = 2KB
  float*          s_sig   = (float*)(smem + 156160);              // 64
  float*          s_lwb   = (float*)(smem + 156416);              // 8

  const v8f vz = VZERO8;
  int tid = threadIdx.x, wave = tid >> 5, lane = tid & 31;
  int n = lane & 15, hi = lane >> 4;
  int row0 = blockIdx.x * 16;

  // one-time constant staging into LDS
  for (int i = tid; i < 2048; i += 256) s_lw[i] = lwW[i];
  for (int i = tid; i < 1024; i += 256) s_bias[i] = bias[i];
  for (int i = tid; i < 512;  i += 256) s_linb[i] = linb[i];
  if (tid < 64) s_sig[tid] = expf(logsig[tid]);
  if (tid < 8)  s_lwb[tid] = lwb[tid];
  for (int idx = tid; idx < 16 * 64; idx += 256) {
    int b = idx >> 6, o = idx & 63;
    s_state[b * 64 + o] = f2bf(state0[(row0 + b) * 64 + o]);
  }
  for (int idx = tid; idx < 16 * 256; idx += 256) {
    int b = idx >> 8, j = idx & 255;
    s_whbf[b * 256 + j] = f2bf(w0[j]);
    s_wc[b * 256 + j] = 0.f;
  }
  // prologue: async-stage u_0 into buffer 0 (2 async instr per wave)
  {
    const unsigned short* usrc = u_bf + ((size_t)row0) * 256;
    for (int idx = tid; idx < 512; idx += 256) {
      int b = idx >> 5, c = (idx & 31) * 8;
      async_ld_lds_b128(&s_u[b * 256 + c], usrc + (size_t)b * 256 + c);
    }
  }

  for (int t = 0; t < 256; ++t) {
    unsigned short* s_ub = s_u + (t & 1) * 4096;   // current u_t buffer
    // issue next step's copies, then wait for the current ones
    if (t < 255) {
      unsigned short* nb = s_u + ((t + 1) & 1) * 4096;
      const unsigned short* usrc = u_bf + ((size_t)(t + 1) * 512 + row0) * 256;
      for (int idx = tid; idx < 512; idx += 256) {
        int b = idx >> 5, c = (idx & 31) * 8;
        async_ld_lds_b128(&nb[b * 256 + c], usrc + (size_t)b * 256 + c);
      }
      wait_async2();                 // only the t+1 batch may remain in flight
    } else {
      wait_async0();
    }
    __syncthreads();

    // ---- LSTM gates = state @ Wih^T + wh @ Whh^T ----
    for (int ti = wave; ti < 64; ti += 8) {
      int col0 = ti * 16;
      __builtin_prefetch(Whh + (size_t)col0 * 256 + 2048, 0, 0);
      v8f acc = vz;
#pragma unroll
      for (int k0 = 0; k0 < 64; k0 += 32)
        acc = wmma_bf16(ld_frag(s_state + k0, 64), ld_frag(Wih + col0 * 64 + k0, 64), acc);
#pragma unroll
      for (int k0 = 0; k0 < 256; k0 += 32)
        acc = wmma_bf16(ld_frag(s_whbf + k0, 256), ld_frag(Whh + col0 * 256 + k0, 256), acc);
#pragma unroll
      for (int v = 0; v < 8; ++v) s_gates[(v + 8 * hi) * 1024 + col0 + n] = acc[v];
    }
    __syncthreads();

    // ---- LSTM nonlinearity (PyTorch gate order i,f,g,o) ----
    for (int idx = tid; idx < 16 * 256; idx += 256) {
      int b = idx >> 8, j = idx & 255;
      const float* g = s_gates + b * 1024;
      float gi = sigmoidf(g[j]       + s_bias[j]);
      float gf = sigmoidf(g[256 + j] + s_bias[256 + j]);
      float gg = tanhf   (g[512 + j] + s_bias[512 + j]);
      float go = sigmoidf(g[768 + j] + s_bias[768 + j]);
      float c  = gf * s_wc[b * 256 + j] + gi * gg;
      s_wc[b * 256 + j]   = c;
      s_whbf[b * 256 + j] = f2bf(go * tanhf(c));
    }
    __syncthreads();

    // ---- mixture logits + softmax ----
    if (tid < 128) {
      int b = tid >> 3, m = tid & 7;
      float acc = s_lwb[m];
      const float* wrow = s_lw + m * 256;
      const unsigned short* hrow = s_whbf + b * 256;
      for (int k = 0; k < 256; ++k) acc += bf2f(hrow[k]) * wrow[k];
      s_w[b * 8 + m] = acc;
    }
    __syncthreads();
    if (tid < 16) {
      float* wr = s_w + tid * 8;
      float mx = wr[0];
#pragma unroll
      for (int m = 1; m < 8; ++m) mx = fmaxf(mx, wr[m]);
      float sum = 0.f;
#pragma unroll
      for (int m = 0; m < 8; ++m) { float e = expf(wr[m] - mx); wr[m] = e; sum += e; }
      float inv = 1.0f / sum;
#pragma unroll
      for (int m = 0; m < 8; ++m) wr[m] *= inv;
    }
    __syncthreads();

    // ---- cand = [state|u_t](16x320) @ linW_flat^T -> (16 x 512) ----
    for (int ti = wave; ti < 32; ti += 8) {
      int col0 = ti * 16;
      __builtin_prefetch(linW + (size_t)col0 * 320 + 2560, 0, 0);
      v8f acc = vz;
#pragma unroll
      for (int k0 = 0; k0 < 320; k0 += 32) {
        v16bf a = (k0 < 64) ? ld_frag(s_state + k0, 64)
                            : ld_frag(s_ub + (k0 - 64), 256);
        acc = wmma_bf16(a, ld_frag(linW + col0 * 320 + k0, 320), acc);
      }
#pragma unroll
      for (int v = 0; v < 8; ++v)
        s_cand[(v + 8 * hi) * 512 + col0 + n] = acc[v] + s_linb[col0 + n];
    }
    __syncthreads();

    // ---- mu = sum_n w[b,n]*cand[b,n,:]; sample; commit new state ----
    for (int idx = tid; idx < 16 * 64; idx += 256) {
      int b = idx >> 6, o = idx & 63;
      const float* wr = s_w + b * 8;
      const float* cr = s_cand + b * 512 + o;
      float mu = 0.f;
#pragma unroll
      for (int m = 0; m < 8; ++m) mu += wr[m] * cr[m * 64];
      int gb = row0 + b;
      unsigned eidx = ((unsigned)t * 512u + (unsigned)gb) * 64u + (unsigned)o;
      float ns = mu + s_sig[o] * hash_normal(eidx, 0x2222u);
      size_t go_ = (size_t)(t * 512 + gb) * 64 + o;
      stf[go_] = ns;
      stb[go_] = f2bf(ns);
      s_state[b * 64 + o] = f2bf(ns);
    }
    __syncthreads();
  }
}

// ---------------------------------------------------------------------------
// Kernel: decoder DBlock + obs sample + x-embedding (time-parallel)
// ---------------------------------------------------------------------------
__global__ void k_decoder(const unsigned short* __restrict__ stb,   // (L*B,64) bf16
                          const unsigned short* __restrict__ W1, const float* __restrict__ b1,
                          const unsigned short* __restrict__ W2, const float* __restrict__ b2,
                          const unsigned short* __restrict__ Wm, const float* __restrict__ bm,
                          const unsigned short* __restrict__ Ws, const float* __restrict__ bs,
                          const unsigned short* __restrict__ xW, const float* __restrict__ xb,
                          float* __restrict__ obs_out,
                          unsigned short* __restrict__ xemb_bf) {
  __shared__ __align__(16) unsigned short s_s[16 * 64];
  __shared__ __align__(16) unsigned short s_t[16 * 256];
  __shared__ __align__(16) float          s_o[16 * 64];
  __shared__ __align__(16) unsigned short s_ob[16 * 32];
  const v8f vz = VZERO8;
  int tid = threadIdx.x, wave = tid >> 5, lane = tid & 31;
  int n = lane & 15, hi = lane >> 4;
  int row0 = blockIdx.x * 16;

  // async-stage the 16x64 bf16 state tile into LDS (128 x 16B)
  if (tid < 128) {
    int b = tid >> 3, c = (tid & 7) * 8;
    async_ld_lds_b128(&s_s[b * 64 + c], stb + (size_t)(row0 + b) * 64 + c);
  }
  wait_async0();
  __syncthreads();

  // trunk: t = tanh(s@W1^T+b1)*sigmoid(s@W2^T+b2), (16 x 256)
  for (int ct = wave; ct < 16; ct += 8) {
    v8f a1 = vz, a2 = vz;
#pragma unroll
    for (int k0 = 0; k0 < 64; k0 += 32) {
      v16bf a = ld_frag(s_s + k0, 64);
      a1 = wmma_bf16(a, ld_frag(W1 + ct * 16 * 64 + k0, 64), a1);
      a2 = wmma_bf16(a, ld_frag(W2 + ct * 16 * 64 + k0, 64), a2);
    }
    int col = ct * 16 + n;
#pragma unroll
    for (int v = 0; v < 8; ++v) {
      int rr = v + 8 * hi;
      s_t[rr * 256 + col] = f2bf(tanhf(a1[v] + b1[col]) * sigmoidf(a2[v] + b2[col]));
    }
  }
  __syncthreads();

  // heads: omu / ols (16 x 32 each)
  if (wave < 4) {
    int mat = wave >> 1, ct = wave & 1;
    const unsigned short* W = mat ? Ws : Wm;
    const float* bb = mat ? bs : bm;
    v8f acc = vz;
#pragma unroll
    for (int k0 = 0; k0 < 256; k0 += 32)
      acc = wmma_bf16(ld_frag(s_t + k0, 256), ld_frag(W + ct * 16 * 256 + k0, 256), acc);
    int col = ct * 16 + n;
#pragma unroll
    for (int v = 0; v < 8; ++v) s_o[(v + 8 * hi) * 64 + mat * 32 + col] = acc[v] + bb[col];
  }
  __syncthreads();

  // sample obs, write fp32 output + bf16 staging
  for (int idx = tid; idx < 512; idx += 256) {
    int b = idx >> 5, c = idx & 31;
    float om = s_o[b * 64 + c], ol = s_o[b * 64 + 32 + c];
    unsigned gi = (unsigned)(row0 + b) * 32u + (unsigned)c;
    float val = om + expf(ol) * hash_normal(gi, 0x3333u);
    obs_out[(size_t)(row0 + b) * 32 + c] = val;
    s_ob[b * 32 + c] = f2bf(val);
  }
  __syncthreads();

  // xemb = tanh(obs @ x_W^T + x_b), (16 x 256) bf16
  for (int ct = wave; ct < 16; ct += 8) {
    v8f acc = wmma_bf16(ld_frag(s_ob, 32), ld_frag(xW + ct * 16 * 32, 32), vz);
    int col = ct * 16 + n;
#pragma unroll
    for (int v = 0; v < 8; ++v)
      xemb_bf[(size_t)(row0 + v + 8 * hi) * 256 + col] = f2bf(tanhf(acc[v] + xb[col]));
  }
}

// ---------------------------------------------------------------------------
// Kernel: posterior LSTM scan over z = [u, xemb]; persistent per batch tile.
// z_t double-buffered with async global->LDS prefetch one step ahead.
// ---------------------------------------------------------------------------
__global__ void k_posterior(const float* __restrict__ h0,
                            const unsigned short* __restrict__ u_bf,
                            const unsigned short* __restrict__ xemb_bf,
                            const unsigned short* __restrict__ Wih,  // (1024,512) bf16
                            const unsigned short* __restrict__ Whh,  // (1024,256) bf16
                            const float* __restrict__ bias,          // 1024
                            float* __restrict__ hf) {
  extern __shared__ char smem[];
  float*          s_gates = (float*)smem;                       // 64KB
  float*          s_wc    = (float*)(smem + 65536);             // 16KB
  unsigned short* s_hbf   = (unsigned short*)(smem + 81920);    // 8KB
  unsigned short* s_z     = (unsigned short*)(smem + 90112);    // 2x16x512 = 32KB
  float*          s_bias  = (float*)(smem + 122880);            // 4KB

  const v8f vz = VZERO8;
  int tid = threadIdx.x, wave = tid >> 5, lane = tid & 31;
  int n = lane & 15, hi = lane >> 4;
  int row0 = blockIdx.x * 16;

  for (int i = tid; i < 1024; i += 256) s_bias[i] = bias[i];
  for (int idx = tid; idx < 16 * 256; idx += 256) {
    int b = idx >> 8, j = idx & 255;
    s_hbf[b * 256 + j] = f2bf(h0[j]);
    s_wc[b * 256 + j] = 0.f;
  }
  // prologue: async-stage z_0 into buffer 0 (4 async instr per wave)
  {
    const unsigned short* ub = u_bf    + ((size_t)row0) * 256;
    const unsigned short* xe = xemb_bf + ((size_t)row0) * 256;
    for (int idx = tid; idx < 1024; idx += 256) {
      int b = idx >> 6, c = (idx & 63) * 8;
      const unsigned short* src = (c < 256) ? (ub + (size_t)b * 256 + c)
                                            : (xe + (size_t)b * 256 + (c - 256));
      async_ld_lds_b128(&s_z[b * 512 + c], src);
    }
  }

  for (int t = 0; t < 256; ++t) {
    unsigned short* s_zb = s_z + (t & 1) * 8192;
    if (t < 255) {
      unsigned short* nb = s_z + ((t + 1) & 1) * 8192;
      const unsigned short* ub = u_bf    + ((size_t)(t + 1) * 512 + row0) * 256;
      const unsigned short* xe = xemb_bf + ((size_t)(t + 1) * 512 + row0) * 256;
      for (int idx = tid; idx < 1024; idx += 256) {
        int b = idx >> 6, c = (idx & 63) * 8;
        const unsigned short* src = (c < 256) ? (ub + (size_t)b * 256 + c)
                                              : (xe + (size_t)b * 256 + (c - 256));
        async_ld_lds_b128(&nb[b * 512 + c], src);
      }
      wait_async4();                 // only the t+1 batch may remain in flight
    } else {
      wait_async0();
    }
    __syncthreads();

    for (int ti = wave; ti < 64; ti += 8) {
      int col0 = ti * 16;
      __builtin_prefetch(Wih + (size_t)col0 * 512 + 4096, 0, 0);
      v8f acc = vz;
#pragma unroll
      for (int k0 = 0; k0 < 512; k0 += 32)
        acc = wmma_bf16(ld_frag(s_zb + k0, 512), ld_frag(Wih + col0 * 512 + k0, 512), acc);
#pragma unroll
      for (int k0 = 0; k0 < 256; k0 += 32)
        acc = wmma_bf16(ld_frag(s_hbf + k0, 256), ld_frag(Whh + col0 * 256 + k0, 256), acc);
#pragma unroll
      for (int v = 0; v < 8; ++v) s_gates[(v + 8 * hi) * 1024 + col0 + n] = acc[v];
    }
    __syncthreads();

    for (int idx = tid; idx < 16 * 256; idx += 256) {
      int b = idx >> 8, j = idx & 255;
      const float* g = s_gates + b * 1024;
      float gi = sigmoidf(g[j]       + s_bias[j]);
      float gf = sigmoidf(g[256 + j] + s_bias[256 + j]);
      float gg = tanhf   (g[512 + j] + s_bias[512 + j]);
      float go = sigmoidf(g[768 + j] + s_bias[768 + j]);
      float c = gf * s_wc[b * 256 + j] + gi * gg;
      float h = go * tanhf(c);
      s_wc[b * 256 + j]  = c;
      s_hbf[b * 256 + j] = f2bf(h);
      if (t == 255) hf[(size_t)(row0 + b) * 256 + j] = h;
    }
    __syncthreads();
  }
}

// ---------------------------------------------------------------------------
// Host launcher
// ---------------------------------------------------------------------------
extern "C" void kernel_launch(void* const* d_in, const int* in_sizes, int n_in,
                              void* d_out, int out_size, void* d_ws, size_t ws_size,
                              hipStream_t stream) {
  (void)in_sizes; (void)n_in; (void)out_size; (void)ws_size;
  const float* ext      = (const float*)d_in[0];
  const float* u_W      = (const float*)d_in[1];
  const float* u_b      = (const float*)d_in[2];
  const float* x_W      = (const float*)d_in[3];
  const float* x_b      = (const float*)d_in[4];
  const float* h0       = (const float*)d_in[5];
  const float* rnn_Wih  = (const float*)d_in[6];
  const float* rnn_Whh  = (const float*)d_in[7];
  const float* rnn_b    = (const float*)d_in[8];
  const float* gd_W1    = (const float*)d_in[9];
  const float* gd_b1    = (const float*)d_in[10];
  const float* gd_W2    = (const float*)d_in[11];
  const float* gd_b2    = (const float*)d_in[12];
  const float* gd_Wm    = (const float*)d_in[13];
  const float* gd_bm    = (const float*)d_in[14];
  const float* gd_Ws    = (const float*)d_in[15];
  const float* gd_bs    = (const float*)d_in[16];
  const float* dyn_w0   = (const float*)d_in[17];
  const float* dyn_Wih  = (const float*)d_in[18];
  const float* dyn_Whh  = (const float*)d_in[19];
  const float* dyn_b    = (const float*)d_in[20];
  const float* dyn_lwW  = (const float*)d_in[21];
  const float* dyn_lwb  = (const float*)d_in[22];
  const float* dyn_linW = (const float*)d_in[23];
  const float* dyn_linb = (const float*)d_in[24];
  const float* dyn_ls   = (const float*)d_in[25];
  const float* dec_W1   = (const float*)d_in[26];
  const float* dec_b1   = (const float*)d_in[27];
  const float* dec_W2   = (const float*)d_in[28];
  const float* dec_b2   = (const float*)d_in[29];
  const float* dec_Wm   = (const float*)d_in[30];
  const float* dec_bm   = (const float*)d_in[31];
  const float* dec_Ws   = (const float*)d_in[32];
  const float* dec_bs   = (const float*)d_in[33];

  char* ws = (char*)d_ws;
  size_t cur = 0;
  auto alloc = [&](size_t bytes) -> void* {
    void* p = ws + cur;
    cur += (bytes + 255) & ~(size_t)255;
    return p;
  };
  const int LB = 256 * 512;  // SEQ * BATCH
  unsigned short* u_bf   = (unsigned short*)alloc((size_t)LB * 256 * 2);
  unsigned short* xe_bf  = (unsigned short*)alloc((size_t)LB * 256 * 2);
  float*          st_f   = (float*)alloc((size_t)LB * 64 * 4);
  unsigned short* st_b   = (unsigned short*)alloc((size_t)LB * 64 * 2);
  float*          state0 = (float*)alloc(512 * 64 * 4);
  unsigned short* b_uW   = (unsigned short*)alloc(8192 * 2);
  unsigned short* b_xW   = (unsigned short*)alloc(8192 * 2);
  unsigned short* b_rWih = (unsigned short*)alloc(524288 * 2);
  unsigned short* b_rWhh = (unsigned short*)alloc(262144 * 2);
  unsigned short* b_dWih = (unsigned short*)alloc(65536 * 2);
  unsigned short* b_dWhh = (unsigned short*)alloc(262144 * 2);
  unsigned short* b_linW = (unsigned short*)alloc(163840 * 2);
  unsigned short* b_eW1  = (unsigned short*)alloc(16384 * 2);
  unsigned short* b_eW2  = (unsigned short*)alloc(16384 * 2);
  unsigned short* b_eWm  = (unsigned short*)alloc(8192 * 2);
  unsigned short* b_eWs  = (unsigned short*)alloc(8192 * 2);

  auto cvt = [&](const float* s, unsigned short* d, int nelem) {
    k_cvt_bf16<<<(nelem + 255) / 256, 256, 0, stream>>>(s, d, nelem);
  };
  cvt(u_W, b_uW, 8192);            cvt(x_W, b_xW, 8192);
  cvt(rnn_Wih, b_rWih, 524288);    cvt(rnn_Whh, b_rWhh, 262144);
  cvt(dyn_Wih, b_dWih, 65536);     cvt(dyn_Whh, b_dWhh, 262144);
  cvt(dyn_linW, b_linW, 163840);
  cvt(dec_W1, b_eW1, 16384);       cvt(dec_W2, b_eW2, 16384);
  cvt(dec_Wm, b_eWm, 8192);        cvt(dec_Ws, b_eWs, 8192);

  k_embed_u<<<LB / 16, 128, 0, stream>>>(ext, b_uW, u_b, u_bf);
  k_state0<<<1, 256, 0, stream>>>(h0, gd_W1, gd_b1, gd_W2, gd_b2,
                                  gd_Wm, gd_bm, gd_Ws, gd_bs, state0);
  k_dynamics<<<512 / 16, 256, 156544, stream>>>(state0, dyn_w0, u_bf, b_dWih, b_dWhh,
                                                dyn_b, dyn_lwW, dyn_lwb, b_linW,
                                                dyn_linb, dyn_ls, st_f, st_b);
  float* out_obs = (float*)d_out;
  float* out_hf  = out_obs + (size_t)LB * 32;
  k_decoder<<<LB / 16, 256, 0, stream>>>(st_b, b_eW1, dec_b1, b_eW2, dec_b2,
                                         b_eWm, dec_bm, b_eWs, dec_bs,
                                         b_xW, x_b, out_obs, xe_bf);
  k_posterior<<<512 / 16, 256, 126976, stream>>>(h0, u_bf, xe_bf, b_rWih, b_rWhh,
                                                 rnn_b, out_hf);
}
